// LSTMPool_8856222564433
// MI455X (gfx1250) — matrix operand order repaired and checked by reading, not compiled, not run on hardware
//
#include <hip/hip_runtime.h>
#include <hip/hip_bf16.h>

// ---------------------------------------------------------------------------
// TimeLSTM (B=2048, L=200, H=128, D=256) + MergeLayer for MI455X (gfx1250).
// All GEMMs use v_wmma_f32_16x16x32_bf16 (bf16 A/B, f32 accumulate).
//   Phase 1: weight f32->bf16 conversion into workspace (L2-resident).
//   Phase 2: u = seq_x @ U_all^T + b_U    (409600x512, K=256, 25600 blocks)
//   Phase 3: 200-step recurrent scan, 16 batch rows per WGP, h/c in LDS,
//            u_t/time_diff software-pipelined across steps (register double
//            buffer) so each serial iteration hides its global latency,
//            fused MergeLayer at the end.
// Workspace requirement: ~840 MB (u buffer) + ~0.5 MB (bf16 weights).
// ---------------------------------------------------------------------------

typedef __attribute__((ext_vector_type(16))) __bf16 v16bf;
typedef __attribute__((ext_vector_type(8)))  float  v8f;

constexpr int Bn = 2048;
constexpr int Ln = 200;
constexpr int Hn = 128;    // hidden
constexpr int G4 = 512;    // 4*H
constexpr int Dn = 256;    // feat + edge
constexpr float INV_SCALE = 1.0f / 5000000.0f;

__device__ inline float sigmoidf_(float x) { return 1.0f / (1.0f + __expf(-x)); }

__device__ inline v8f wmma_bf16(v16bf a, v16bf b, v8f c) {
    return __builtin_amdgcn_wmma_f32_16x16x32_bf16(
        /*neg_a=*/false, a, /*neg_b=*/false, b,
        /*c_mod=*/(short)0, c, /*reuse_a=*/false, /*reuse_b=*/false);
}

// ---- fragment loaders (A/B in contiguous k order k = 16*(lane/16) + s, wave32) --------------
// A (16xK window [k0,k0+32), 16-bit): lane l<16 -> row l, K k0..k0+15
//                                     lane l>=16 -> row l-16, K k0+16..k0+31
__device__ inline v16bf a_frag_f32(const float* a, int ld, int k0) {
    int lane = threadIdx.x & 31;
    int r    = lane & 15;
    int kh   = k0 + ((lane & 16) ? 16 : 0);
    const float* p = a + (size_t)r * ld + kh;
    v16bf o;
#pragma unroll
    for (int s = 0; s < 8; ++s) o[s]     = (__bf16)p[s];
#pragma unroll
    for (int s = 0; s < 8; ++s) o[s + 8] = (__bf16)p[s + 8];
    return o;
}

// B (KxN, K window [k0,k0+32)) from row-major bf16 weight W[n][k]  (B = W^T):
// lane n<16 -> col n, K k0..k0+15 ; lane>=16 -> col n-16, K k0+16..k0+31.
// 16 consecutive bf16 = 32B aligned -> single v16bf load.
__device__ inline v16bf b_frag_bf16(const __bf16* w, int ld, int n0, int k0) {
    int lane = threadIdx.x & 31;
    const __bf16* p = w + (size_t)(n0 + (lane & 15)) * ld + k0 + ((lane & 16) ? 16 : 0);
    return *reinterpret_cast<const v16bf*>(p);
}

// C/D layout: element v of v8f -> row ((lane&16)?8:0)+v, col lane&15.
__device__ inline int cd_row(int v) { return ((threadIdx.x & 16) ? 8 : 0) + v; }
__device__ inline int cd_col()      { return threadIdx.x & 15; }

// ---- Phase 1: f32 -> bf16 weight conversion --------------------------------
__global__ void cvt_f32_bf16_kernel(const float* __restrict__ s,
                                    __bf16* __restrict__ d, int n) {
    int i = blockIdx.x * blockDim.x + threadIdx.x;
    if (i < n) d[i] = (__bf16)s[i];
}

// ---- Phase 2: u = seq_x @ U_all^T + b_U ------------------------------------
// grid = (B*L)/16 blocks, 256 threads (8 waves); wave w -> cols [64w, 64w+64)
__global__ __launch_bounds__(256)
void u_proj_kernel(const float* __restrict__ seq, const float* __restrict__ seq_e,
                   const __bf16* __restrict__ Ubf, const float* __restrict__ bU,
                   float* __restrict__ u) {
    int row0 = blockIdx.x * 16;          // flat row index = b*L + l
    int wv   = threadIdx.x >> 5;

    v8f acc[4];
#pragma unroll
    for (int t = 0; t < 4; ++t) {
        float bv = bU[wv * 64 + t * 16 + cd_col()];
#pragma unroll
        for (int v = 0; v < 8; ++v) acc[t][v] = bv;
    }

    const float* a0 = seq   + (size_t)row0 * Hn;
    const float* a1 = seq_e + (size_t)row0 * Hn;
#pragma unroll 1
    for (int ks = 0; ks < 8; ++ks) {     // K = 256 in 8 steps of 32
        v16bf a = a_frag_f32(ks < 4 ? a0 : a1, Hn, (ks & 3) * 32);
#pragma unroll
        for (int t = 0; t < 4; ++t) {
            v16bf b = b_frag_bf16(Ubf, Dn, wv * 64 + t * 16, ks * 32);
            acc[t] = wmma_bf16(a, b, acc[t]);
        }
    }
#pragma unroll
    for (int t = 0; t < 4; ++t) {
        int n = wv * 64 + t * 16 + cd_col();
#pragma unroll
        for (int v = 0; v < 8; ++v)
            u[(size_t)(row0 + cd_row(v)) * G4 + n] = acc[t][v];
    }
}

// ---- Phase 3: recurrent scan + fused MergeLayer ----------------------------
// grid = B/16 = 128 blocks, 256 threads (8 waves). 16 batch rows per block.
// u_t and time_diff are register double-buffered: loads for step l+1 are
// issued before the WMMA phase of step l, so the s_wait lands a full
// iteration (20 WMMAs + 2 barriers + elementwise) later.
__global__ __launch_bounds__(256)
void scan_kernel(const float* __restrict__ u, const float* __restrict__ td,
                 const __bf16* __restrict__ Wallbf, const float* __restrict__ bW,
                 const __bf16* __restrict__ Wdbf,   const float* __restrict__ bd,
                 const float* __restrict__ src,
                 const __bf16* __restrict__ fc1bf,  const float* __restrict__ fc1b,
                 const __bf16* __restrict__ fc2bf,  const float* __restrict__ fc2b,
                 float* __restrict__ out) {
    __shared__ float sH[16 * Hn];     // h (f32)
    __shared__ float sC[16 * Hn];     // c (f32)
    __shared__ float sAux[16 * Hn];   // c_s1 pre-tanh; later src tile
    __shared__ float sG[16 * G4];     // gates; later hmid

    int r0 = blockIdx.x * 16;
    int wv = threadIdx.x >> 5;
    int em = threadIdx.x >> 4;        // elementwise row 0..15
    int ej0 = (threadIdx.x & 15) * 8; // elementwise col base

    for (int i = threadIdx.x; i < 16 * Hn; i += 256) { sH[i] = 0.f; sC[i] = 0.f; }

    // preload step 0: u_t tile slots for this wave + time_diff
    v8f ucur[4];
#pragma unroll
    for (int t = 0; t < 4; ++t) {
        int n = wv * 64 + t * 16 + cd_col();
#pragma unroll
        for (int v = 0; v < 8; ++v)
            ucur[t][v] = u[((size_t)(r0 + cd_row(v)) * Ln + 0) * G4 + n];
    }
    float tdl_c = td[(size_t)(r0 + em) * Ln + 0];
    float tdp_c = tdl_c;              // span(0) == 0
    __syncthreads();

#pragma unroll 1
    for (int l = 0; l < Ln; ++l) {
        // -- issue next-step loads first (consumed next iteration) --
        v8f unext[4];
        float tdl_n = 0.f, tdp_n = 0.f;
        if (l + 1 < Ln) {
#pragma unroll
            for (int t = 0; t < 4; ++t) {
                int n = wv * 64 + t * 16 + cd_col();
#pragma unroll
                for (int v = 0; v < 8; ++v)
                    unext[t][v] = u[((size_t)(r0 + cd_row(v)) * Ln + (l + 1)) * G4 + n];
            }
            tdp_n = tdl_c;                                   // td[l]
            tdl_n = td[(size_t)(r0 + em) * Ln + (l + 1)];    // td[l+1]
        }
        if (l + 2 < Ln) {   // keep the HBM stream warm in L2
            const float* p = u + ((size_t)(r0 + (threadIdx.x & 15)) * Ln + (l + 2)) * G4
                               + wv * 64;
            __builtin_prefetch(p, 0, 3);
        }

        // -- c_s1 accumulation: one 16x16 tile per wave (cols wv*16..+16)
        {
            v8f acc;
            float bv = bd[wv * 16 + cd_col()];
#pragma unroll
            for (int v = 0; v < 8; ++v) acc[v] = bv;
#pragma unroll
            for (int ks = 0; ks < 4; ++ks) {
                v16bf a = a_frag_f32(sC, Hn, ks * 32);
                v16bf b = b_frag_bf16(Wdbf, Hn, wv * 16, ks * 32);
                acc = wmma_bf16(a, b, acc);
            }
#pragma unroll
            for (int v = 0; v < 8; ++v)
                sAux[cd_row(v) * Hn + wv * 16 + cd_col()] = acc[v];
        }
        // -- gates = h@W_all^T + b_W + u_t : 4 tiles per wave (cols wv*64..+64)
        {
            v8f acc[4];
#pragma unroll
            for (int t = 0; t < 4; ++t) {
                float bv = bW[wv * 64 + t * 16 + cd_col()];
#pragma unroll
                for (int v = 0; v < 8; ++v) acc[t][v] = bv + ucur[t][v];
            }
#pragma unroll
            for (int ks = 0; ks < 4; ++ks) {
                v16bf a = a_frag_f32(sH, Hn, ks * 32);
#pragma unroll
                for (int t = 0; t < 4; ++t) {
                    v16bf b = b_frag_bf16(Wallbf, Hn, wv * 64 + t * 16, ks * 32);
                    acc[t] = wmma_bf16(a, b, acc[t]);
                }
            }
#pragma unroll
            for (int t = 0; t < 4; ++t) {
                int n = wv * 64 + t * 16 + cd_col();
#pragma unroll
                for (int v = 0; v < 8; ++v) sG[cd_row(v) * G4 + n] = acc[t][v];
            }
        }
        __syncthreads();

        // -- elementwise gate update: thread t -> row t>>4, cols (t&15)*8..+8
        {
            float span = (tdp_c - tdl_c) * INV_SCALE;
#pragma unroll
            for (int q = 0; q < 8; ++q) {
                int j = ej0 + q;
                int idx = em * Hn + j;
                float c_old = sC[idx];
                float cs1   = tanhf(sAux[idx]);
                float c_adj = c_old - cs1 + cs1 * span;
                float f  = sigmoidf_(sG[em * G4 + j]);
                float ig = sigmoidf_(sG[em * G4 + 128 + j]);
                float o  = sigmoidf_(sG[em * G4 + 256 + j]);
                float g  = sigmoidf_(sG[em * G4 + 384 + j]);  // repo uses sigmoid
                float c_new = f * c_adj + ig * g;
                sC[idx] = c_new;
                sH[idx] = o * tanhf(c_new);
            }
        }
        __syncthreads();

        // rotate double buffers
#pragma unroll
        for (int t = 0; t < 4; ++t) ucur[t] = unext[t];
        tdl_c = tdl_n;
        tdp_c = tdp_n;
    }

    // ---- fused MergeLayer: out = fc2( relu( fc1( [hn, src] ) ) ) ----
    for (int i = threadIdx.x; i < 16 * Hn; i += 256)
        sAux[i] = src[(size_t)(r0 + (i >> 7)) * Hn + (i & 127)];
    __syncthreads();

    {   // hmid: one 16x16 tile per wave, K=256 (h then src)
        v8f acc;
        float bv = fc1b[wv * 16 + cd_col()];
#pragma unroll
        for (int v = 0; v < 8; ++v) acc[v] = bv;
#pragma unroll
        for (int ks = 0; ks < 8; ++ks) {
            const float* A = (ks < 4) ? sH : sAux;
            v16bf a = a_frag_f32(A, Hn, (ks & 3) * 32);
            v16bf b = b_frag_bf16(fc1bf, Dn, wv * 16, ks * 32);
            acc = wmma_bf16(a, b, acc);
        }
#pragma unroll
        for (int v = 0; v < 8; ++v)
            sG[cd_row(v) * Hn + wv * 16 + cd_col()] = fmaxf(acc[v], 0.f);
    }
    __syncthreads();
    {   // out: one 16x16 tile per wave, K=128
        v8f acc;
        float bv = fc2b[wv * 16 + cd_col()];
#pragma unroll
        for (int v = 0; v < 8; ++v) acc[v] = bv;
#pragma unroll
        for (int ks = 0; ks < 4; ++ks) {
            v16bf a = a_frag_f32(sG, Hn, ks * 32);
            v16bf b = b_frag_bf16(fc2bf, Hn, wv * 16, ks * 32);
            acc = wmma_bf16(a, b, acc);
        }
#pragma unroll
        for (int v = 0; v < 8; ++v)
            out[(size_t)(r0 + cd_row(v)) * Hn + wv * 16 + cd_col()] = acc[v];
    }
}

// ---------------------------------------------------------------------------
extern "C" void kernel_launch(void* const* d_in, const int* in_sizes, int n_in,
                              void* d_out, int out_size, void* d_ws, size_t ws_size,
                              hipStream_t stream) {
    const float* src   = (const float*)d_in[0];
    const float* seq   = (const float*)d_in[2];
    const float* seq_e = (const float*)d_in[4];
    const float* td    = (const float*)d_in[5];
    const float* W_all = (const float*)d_in[7];
    const float* b_W   = (const float*)d_in[8];
    const float* U_all = (const float*)d_in[9];
    const float* b_U   = (const float*)d_in[10];
    const float* W_d   = (const float*)d_in[11];
    const float* b_d   = (const float*)d_in[12];
    const float* fc1_w = (const float*)d_in[13];
    const float* fc1_b = (const float*)d_in[14];
    const float* fc2_w = (const float*)d_in[15];
    const float* fc2_b = (const float*)d_in[16];
    float* out = (float*)d_out;

    // workspace layout
    char* ws = (char*)d_ws;
    float* u = (float*)ws;
    size_t off = (size_t)Bn * Ln * G4 * sizeof(float);          // 838,860,800 B
    __bf16* Ubf    = (__bf16*)(ws + off); off += (size_t)G4 * Dn * 2;
    __bf16* Wallbf = (__bf16*)(ws + off); off += (size_t)G4 * Hn * 2;
    __bf16* Wdbf   = (__bf16*)(ws + off); off += (size_t)Hn * Hn * 2;
    __bf16* fc1bf  = (__bf16*)(ws + off); off += (size_t)Hn * Dn * 2;
    __bf16* fc2bf  = (__bf16*)(ws + off); off += (size_t)Hn * Hn * 2;

    auto cvt = [&](const float* s, __bf16* d, int n) {
        cvt_f32_bf16_kernel<<<(n + 255) / 256, 256, 0, stream>>>(s, d, n);
    };
    cvt(U_all, Ubf,    G4 * Dn);
    cvt(W_all, Wallbf, G4 * Hn);
    cvt(W_d,   Wdbf,   Hn * Hn);
    cvt(fc1_w, fc1bf,  Hn * Dn);
    cvt(fc2_w, fc2bf,  Hn * Hn);

    u_proj_kernel<<<(Bn * Ln) / 16, 256, 0, stream>>>(seq, seq_e, Ubf, b_U, u);

    scan_kernel<<<Bn / 16, 256, 0, stream>>>(u, td, Wallbf, b_W, Wdbf, b_d,
                                             src, fc1bf, fc1_b, fc2bf, fc2_b, out);
}
